// sLSTMblock_37735582663289
// MI455X (gfx1250) — matrix operand order repaired and checked by reading, not compiled, not run on hardware
//
#include <hip/hip_runtime.h>
#include <cstdint>
#include <cstddef>

#define B_  32
#define T_  2048
#define D_  512
#define H_  512
#define G4_ 2048   // 4*H

typedef __attribute__((ext_vector_type(16))) __bf16 v16bf;
typedef __attribute__((ext_vector_type(8)))  float  v8f;
typedef int v4i __attribute__((vector_size(16)));   // matches builtin prototype

union Frag {
  v16bf bf;
  uint4 q[2];
};

#define HAS_ASYNC_LDS (__has_builtin(__builtin_amdgcn_global_load_async_to_lds_b128) && \
                       __has_builtin(__builtin_amdgcn_s_wait_asynccnt))

__device__ __forceinline__ unsigned short f32_to_bf16_rne(float f) {
  union { float f; unsigned int u; } v; v.f = f;
  unsigned int u = v.u;
  unsigned int r = u + 0x7FFFu + ((u >> 16) & 1u);
  return (unsigned short)(r >> 16);
}

// ---------------- fp32 -> bf16 conversion ----------------
__global__ void cvt_f32_bf16(const float* __restrict__ in,
                             unsigned short* __restrict__ out, int n) {
  for (int i = blockIdx.x * blockDim.x + threadIdx.x; i < n;
       i += gridDim.x * blockDim.x)
    out[i] = f32_to_bf16_rne(in[i]);
}

// ---------------- per-launch state init (ws is poisoned) ----------------
__global__ void init_state(unsigned short* __restrict__ hbufs,
                           int* __restrict__ bar) {
  int i = blockIdx.x * blockDim.x + threadIdx.x;
  if (i < 2 * B_ * H_) hbufs[i] = 0;          // h double buffer = 0
  if (i < 2) bar[i] = 0;                      // barrier count + generation
}

// ---------------- Phase 1: wx = x @ W^T + W_b  (bf16 WMMA) ----------------
// Register-blocked: each wave computes a 32x32 output (4 accumulators), so each
// A/B fragment feeds two WMMAs (2x data reuse vs one-tile-per-wave).
// grid: (BT/32, 4H/(32*4)), block: 128 (4 waves; wave = one 32-wide N block)
__global__ void __launch_bounds__(128)
wx_gemm(const unsigned short* __restrict__ Xb,   // [B*T, D] bf16
        const unsigned short* __restrict__ Wb,   // [4H, D]  bf16
        const float* __restrict__ Wbias,         // [4H]
        float* __restrict__ wx) {                // [B*T, 4H] fp32
  const int lane = threadIdx.x & 31;
  const int wave = threadIdx.x >> 5;
  const int row  = lane & 15;
  const int hi   = lane >> 4;
  const int m0   = blockIdx.x * 32;
  const int n0   = (blockIdx.y * 4 + wave) * 32;

  v8f accAA = {0.f,0.f,0.f,0.f,0.f,0.f,0.f,0.f};
  v8f accAB = accAA, accBA = accAA, accBB = accAA;

  const uint4* Arow0 = (const uint4*)(Xb + (size_t)(m0 + row) * D_);
  const uint4* Arow1 = (const uint4*)(Xb + (size_t)(m0 + 16 + row) * D_);
  const uint4* Brow0 = (const uint4*)(Wb + (size_t)(n0 + row) * D_);
  const uint4* Brow1 = (const uint4*)(Wb + (size_t)(n0 + 16 + row) * D_);
#pragma unroll 4
  for (int k0 = 0; k0 < D_; k0 += 32) {
    Frag a0, a1, b0, b1;
    // A 16x32 bf16: lane holds row, K = k0 + hi*8 + {0..7, 16..23}
    a0.q[0] = Arow0[(k0 + hi * 8) >> 3];
    a0.q[1] = Arow0[(k0 + hi * 8 + 16) >> 3];
    a1.q[0] = Arow1[(k0 + hi * 8) >> 3];
    a1.q[1] = Arow1[(k0 + hi * 8 + 16) >> 3];
    // B 32x16 bf16: lane holds col, K = k0 + hi*8 + {0..7, 16..23}
    b0.q[0] = Brow0[(k0 + hi * 8) >> 3];
    b0.q[1] = Brow0[(k0 + hi * 8 + 16) >> 3];
    b1.q[0] = Brow1[(k0 + hi * 8) >> 3];
    b1.q[1] = Brow1[(k0 + hi * 8 + 16) >> 3];
    accAA = __builtin_amdgcn_wmma_f32_16x16x32_bf16(false, a0.bf, false, b0.bf,
                                                    (short)0, accAA, false, false);
    accAB = __builtin_amdgcn_wmma_f32_16x16x32_bf16(false, a0.bf, false, b1.bf,
                                                    (short)0, accAB, false, false);
    accBA = __builtin_amdgcn_wmma_f32_16x16x32_bf16(false, a1.bf, false, b0.bf,
                                                    (short)0, accBA, false, false);
    accBB = __builtin_amdgcn_wmma_f32_16x16x32_bf16(false, a1.bf, false, b1.bf,
                                                    (short)0, accBB, false, false);
  }
  const float biasA = Wbias[n0 + row];
  const float biasB = Wbias[n0 + 16 + row];
#pragma unroll
  for (int r = 0; r < 8; ++r) {
    int mA = m0 + r + hi * 8;             // C/D layout: M = r + 8*hi, N = row
    int mB = mA + 16;
    wx[(size_t)mA * G4_ + n0 + row]      = accAA[r] + biasA;
    wx[(size_t)mA * G4_ + n0 + 16 + row] = accAB[r] + biasB;
    wx[(size_t)mB * G4_ + n0 + row]      = accBA[r] + biasA;
    wx[(size_t)mB * G4_ + n0 + 16 + row] = accBB[r] + biasB;
  }
}

// ---------------- Phase 2: persistent recurrent scan ----------------
// 32 workgroups x 128 threads. Block b owns h columns [16b, 16b+16).
// Wave w computes gate w (i/f/o/g) for its 16 columns, both 16-row batch tiles.
// Per step: async-stage h (32 KB) into LDS once, WMMA against the LDS-resident
// 64 KB U slice, then do the elementwise c/h update locally.
__global__ void __launch_bounds__(128)
slstm_scan(const unsigned short* __restrict__ Ub16,  // [4H, H] bf16
           const float* __restrict__ Ubias,          // [4H]
           const float* __restrict__ alpha,          // [H]
           const float* __restrict__ wx,             // [B*T, 4H]
           unsigned short* __restrict__ hbufs,       // 2 x [B, H] bf16
           int* __restrict__ bar,                    // {count, generation}
           float* __restrict__ out) {                // outs [B,T,H] ++ h ++ c
  extern __shared__ char smem[];
  unsigned short* Us = (unsigned short*)smem;                      // 64x512 bf16 = 64 KB
  unsigned short* hS = (unsigned short*)(smem + 64 * D_ * 2);      // 32x512 bf16 = 32 KB
  float* gatesL = (float*)(smem + 64 * D_ * 2 + B_ * H_ * 2);      // [4][32][16] = 8 KB
  float* cL     = gatesL + 4 * B_ * 16;                            // [32*16] = 2 KB

  const int tid  = threadIdx.x;
  const int lane = tid & 31;
  const int wave = tid >> 5;     // gate index 0..3
  const int row  = lane & 15;
  const int hi   = lane >> 4;
  const int hcol0 = blockIdx.x * 16;

  // Preload this WG's 64 U rows (gate*512 + hcol0 + j) into LDS.
  {
    const uint4* src = (const uint4*)Ub16;
    uint4* dst = (uint4*)Us;
    for (int i = tid; i < 64 * (D_ / 8); i += 128) {
      int r  = i >> 6;                       // local row 0..63
      int cw = i & 63;                       // uint4 within row
      int grow = (r >> 4) * H_ + hcol0 + (r & 15);
      dst[i] = src[(size_t)grow * (D_ / 8) + cw];
    }
    for (int i = tid; i < B_ * 16; i += 128) cL[i] = 0.f;
  }
  __syncthreads();

  const float ub = Ubias[wave * H_ + hcol0 + row];
  const uint4* UsQ = (const uint4*)Us;
  const uint4* hQ  = (const uint4*)hS;
  const int usbase = ((wave * 16 + row) * D_) >> 3;

  for (int t = 0; t < T_; ++t) {
    // ---- stage h(t) into LDS: 2048 x 16B chunks over 128 threads ----
    const unsigned short* hsrc = hbufs + (size_t)(t & 1) * B_ * H_;
#if HAS_ASYNC_LDS
    for (int i = tid; i < (B_ * H_ * 2) / 16; i += 128) {
      __builtin_amdgcn_global_load_async_to_lds_b128(
          (v4i*)((const char*)hsrc + i * 16),   // global src (cast drops const)
          (v4i*)((char*)hS + i * 16),           // LDS dst
          0, 0);
    }
    __builtin_amdgcn_s_wait_asynccnt(0);
#else
    for (int i = tid; i < (B_ * H_ * 2) / 16; i += 128)
      ((uint4*)hS)[i] = ((const uint4*)hsrc)[i];
#endif
    __syncthreads();

    v8f acc0 = {0.f,0.f,0.f,0.f,0.f,0.f,0.f,0.f};
    v8f acc1 = acc0;
#pragma unroll 4
    for (int k0 = 0; k0 < H_; k0 += 32) {
      Frag bfr, a0, a1;
      bfr.q[0] = UsQ[usbase + ((k0 + hi * 8) >> 3)];         // ds_load_b128
      bfr.q[1] = UsQ[usbase + ((k0 + hi * 8 + 16) >> 3)];
      a0.q[0] = hQ[((row)      * H_ + k0 + hi * 8) >> 3];    // ds_load_b128
      a0.q[1] = hQ[((row)      * H_ + k0 + hi * 8 + 16) >> 3];
      a1.q[0] = hQ[((16 + row) * H_ + k0 + hi * 8) >> 3];
      a1.q[1] = hQ[((16 + row) * H_ + k0 + hi * 8 + 16) >> 3];
      acc0 = __builtin_amdgcn_wmma_f32_16x16x32_bf16(false, a0.bf, false, bfr.bf,
                                                     (short)0, acc0, false, false);
      acc1 = __builtin_amdgcn_wmma_f32_16x16x32_bf16(false, a1.bf, false, bfr.bf,
                                                     (short)0, acc1, false, false);
    }
    // gates = wmma + U_b + wx_t  -> LDS
#pragma unroll
    for (int r = 0; r < 8; ++r) {
      int mA = r + hi * 8;
      int mB = 16 + mA;
      size_t wA = ((size_t)mA * T_ + t) * G4_ + wave * H_ + hcol0 + row;
      size_t wB = ((size_t)mB * T_ + t) * G4_ + wave * H_ + hcol0 + row;
      gatesL[(wave * B_ + mA) * 16 + row] = acc0[r] + ub + wx[wA];
      gatesL[(wave * B_ + mB) * 16 + row] = acc1[r] + ub + wx[wB];
    }
    __syncthreads();

    unsigned short* hnext = hbufs + (size_t)((t + 1) & 1) * B_ * H_;
    for (int e = tid; e < B_ * 16; e += 128) {
      int m = e >> 4, n = e & 15;
      float gi = gatesL[(0 * B_ + m) * 16 + n];
      float gf = gatesL[(1 * B_ + m) * 16 + n];
      float go = gatesL[(2 * B_ + m) * 16 + n];
      float gg = gatesL[(3 * B_ + m) * 16 + n];
      float i_ = 1.f / (1.f + __expf(-gi));
      float f_ = 1.f / (1.f + __expf(-gf));
      float o_ = 1.f / (1.f + __expf(-go));
      float g_ = tanhf(gg);
      float al = alpha[hcol0 + n];
      float cc = al * (f_ * cL[e] + i_ * g_);
      cL[e] = cc;
      float hh = o_ * tanhf(cc);
      out[((size_t)m * T_ + t) * H_ + hcol0 + n] = hh;
      hnext[(size_t)m * H_ + hcol0 + n] = f32_to_bf16_rne(hh);
      if (t == T_ - 1) {
        out[(size_t)B_ * T_ * H_ + (size_t)m * H_ + hcol0 + n] = hh;
        out[(size_t)B_ * T_ * H_ + (size_t)B_ * H_ + (size_t)m * H_ + hcol0 + n] = cc;
      }
    }

    // One grid-wide barrier per timestep across the 32 persistent WGs.
    __threadfence();
    __syncthreads();
    if (tid == 0) {
      int g = __hip_atomic_load(&bar[1], __ATOMIC_ACQUIRE, __HIP_MEMORY_SCOPE_AGENT);
      if (__hip_atomic_fetch_add(&bar[0], 1, __ATOMIC_ACQ_REL,
                                 __HIP_MEMORY_SCOPE_AGENT) == 31) {
        __hip_atomic_store(&bar[0], 0, __ATOMIC_RELAXED, __HIP_MEMORY_SCOPE_AGENT);
        __hip_atomic_store(&bar[1], g + 1, __ATOMIC_RELEASE, __HIP_MEMORY_SCOPE_AGENT);
      } else {
        while (__hip_atomic_load(&bar[1], __ATOMIC_ACQUIRE,
                                 __HIP_MEMORY_SCOPE_AGENT) == g)
          __builtin_amdgcn_s_sleep(1);
      }
    }
    __syncthreads();
  }
}

// ---------------- host side ----------------
extern "C" void kernel_launch(void* const* d_in, const int* in_sizes, int n_in,
                              void* d_out, int out_size, void* d_ws, size_t ws_size,
                              hipStream_t stream) {
  const float* x     = (const float*)d_in[0];   // [B,T,D]
  const float* W_w   = (const float*)d_in[1];   // [4H,D]
  const float* W_b   = (const float*)d_in[2];   // [4H]
  const float* U_w   = (const float*)d_in[3];   // [4H,H]
  const float* U_b   = (const float*)d_in[4];   // [4H]
  const float* alpha = (const float*)d_in[5];   // [H]
  float* out = (float*)d_out;

  char* ws = (char*)d_ws;
  int* bar = (int*)ws;                                       // 256 B reserved
  unsigned short* hbufs = (unsigned short*)(ws + 256);       // 64 KB (2 x B*H bf16)
  unsigned short* Xb = (unsigned short*)(ws + 256 + 65536);  // 64 MB
  unsigned short* Wb = Xb + (size_t)B_ * T_ * D_;            // 2 MB
  unsigned short* Ub = Wb + (size_t)G4_ * D_;                // 2 MB
  float* wx = (float*)(Ub + (size_t)G4_ * H_);               // 512 MB

  cvt_f32_bf16<<<8192, 256, 0, stream>>>(x,   Xb, B_ * T_ * D_);
  cvt_f32_bf16<<<1024, 256, 0, stream>>>(W_w, Wb, G4_ * D_);
  cvt_f32_bf16<<<1024, 256, 0, stream>>>(U_w, Ub, G4_ * H_);
  init_state<<<128, 256, 0, stream>>>(hbufs, bar);

  dim3 g1(B_ * T_ / 32, (G4_ / 16) / 8);                     // (2048, 16)
  wx_gemm<<<g1, 128, 0, stream>>>(Xb, Wb, W_b, wx);

  size_t lds = (size_t)64 * D_ * 2 + (size_t)B_ * H_ * 2 +
               4 * B_ * 16 * 4 + B_ * 16 * 4;                // 108,544 B
  slstm_scan<<<32, 128, lds, stream>>>(Ub, U_b, alpha, wx, hbufs, bar, out);
}
